// Qwen3MoeAfdMlpLayer_22874995818758
// MI455X (gfx1250) — matrix operand order repaired and checked by reading, not compiled, not run on hardware
//
#include <hip/hip_runtime.h>

// ---------------- problem constants (from reference) ----------------
#define Hdim 2048
#define Idim 768
#define NEXP 64
#define TOPK 8

// ---------------- tiling ----------------
#define TM 32                  // tokens per block (two 16-row WMMA tiles)
#define KC 256                 // K-chunk of H staged into LDS (f32 via async DMA)
#define XSF 268                // dword stride of staged X chunk (268%64=12 -> conflict-free)
#define HS (Idim + 8)          // ushort stride for h matrix (bank-conflict-free)
#define NTHREADS 512           // 16 wave32
#define NWAVE 16
#define GT1 3                  // Idim / (16*NWAVE): gate tiles per wave (and 3 up tiles)
#define NT2 8                  // Hdim / (16*NWAVE): output tiles per wave

typedef __attribute__((ext_vector_type(16))) __bf16 v16bf;
typedef __attribute__((ext_vector_type(8)))  float  v8f;

// Pack two f32 -> two bf16 (truncate) in ONE v_perm_b32.
// dst = { hi16(b) , hi16(a) }  (a -> low half = even K index)
__device__ __forceinline__ unsigned pku(unsigned a, unsigned b) {
    return __builtin_amdgcn_perm(b, a, 0x07060302u);
}
__device__ __forceinline__ unsigned pk(float a, float b) {
    return pku(__float_as_uint(a), __float_as_uint(b));
}
__device__ __forceinline__ unsigned short f2bf1(float f) {
    return (unsigned short)(__float_as_uint(f) >> 16);
}

struct BF { union { v16bf v; unsigned u[8]; uint4 q[2]; }; };

// B fragment (32x16 bf16): this lane = one column n; K half selected by caller.
// rowptr points at W[n*ld + kb]; f32 kb..kb+7 and kb+16..kb+23 -> 4x b128 loads + 8 perms.
__device__ __forceinline__ BF load_b_f32(const float* __restrict__ rowptr) {
    BF b;
    const float4* p = (const float4*)rowptr;
    float4 f0 = p[0], f1 = p[1], f2 = p[4], f3 = p[5];
    b.u[0] = pk(f0.x, f0.y); b.u[1] = pk(f0.z, f0.w);
    b.u[2] = pk(f1.x, f1.y); b.u[3] = pk(f1.z, f1.w);
    b.u[4] = pk(f2.x, f2.y); b.u[5] = pk(f2.z, f2.w);
    b.u[6] = pk(f3.x, f3.y); b.u[7] = pk(f3.z, f3.w);
    return b;
}

// A fragment (16x32 bf16) from f32 data in LDS: K k0..k0+7 and k0+16..k0+23
__device__ __forceinline__ BF load_a_f32lds(const unsigned* rowbase, int k0) {
    uint4 q0 = *(const uint4*)(rowbase + k0);
    uint4 q1 = *(const uint4*)(rowbase + k0 + 4);
    uint4 q2 = *(const uint4*)(rowbase + k0 + 16);
    uint4 q3 = *(const uint4*)(rowbase + k0 + 20);
    BF a;
    a.u[0] = pku(q0.x, q0.y); a.u[1] = pku(q0.z, q0.w);
    a.u[2] = pku(q1.x, q1.y); a.u[3] = pku(q1.z, q1.w);
    a.u[4] = pku(q2.x, q2.y); a.u[5] = pku(q2.z, q2.w);
    a.u[6] = pku(q3.x, q3.y); a.u[7] = pku(q3.z, q3.w);
    return a;
}

// A fragment (16x32 bf16) from bf16 data in LDS (16B-aligned)
__device__ __forceinline__ BF load_a_lds(const unsigned short* rowbase, int k0) {
    BF a;
    a.q[0] = *(const uint4*)(rowbase + k0);
    a.q[1] = *(const uint4*)(rowbase + k0 + 16);
    return a;
}

__device__ __forceinline__ v8f wmma_bf16(const BF& a, const BF& b, v8f c) {
    return __builtin_amdgcn_wmma_f32_16x16x32_bf16(false, a.v, false, b.v,
                                                   (short)0, c, false, false);
}

// ---- CDNA5 async global->LDS DMA (ASYNCcnt-tracked), 16B per lane ----
__device__ __forceinline__ void async_copy16(unsigned lds_byte_addr, const void* gaddr) {
    asm volatile("global_load_async_to_lds_b128 %0, %1, off"
                 :: "v"(lds_byte_addr), "v"((unsigned long long)(size_t)gaddr)
                 : "memory");
}
__device__ __forceinline__ void wait_async0() {
    asm volatile("s_wait_asynccnt 0x0" ::: "memory");
}

// ---------------- routing kernels ----------------
__global__ void k_count(const int* __restrict__ ids, int n, int* __restrict__ cnt) {
    int i = blockIdx.x * blockDim.x + threadIdx.x;
    if (i < n) atomicAdd(&cnt[ids[i]], 1);
}

__global__ void k_scan(const int* __restrict__ cnt, int* __restrict__ offs,
                       int* __restrict__ toffs) {
    if (threadIdx.x == 0 && blockIdx.x == 0) {
        int a = 0, t = 0;
        for (int e = 0; e < NEXP; e++) {
            offs[e] = a; toffs[e] = t;
            a += cnt[e]; t += (cnt[e] + TM - 1) / TM;
        }
        offs[NEXP] = a; toffs[NEXP] = t;
    }
}

__global__ void k_fill(const int* __restrict__ ids, const float* __restrict__ wts,
                       int n, const int* __restrict__ offs, int* __restrict__ fill,
                       int* __restrict__ tok, float* __restrict__ cww) {
    int i = blockIdx.x * blockDim.x + threadIdx.x;
    if (i < n) {
        int e = ids[i];
        int p = offs[e] + atomicAdd(&fill[e], 1);
        tok[p] = i / TOPK;
        cww[p] = wts[i];
    }
}

// ---------------- fused grouped GEMM1 + SiLU*up + GEMM2 ----------------
__global__ __launch_bounds__(NTHREADS, 1)
void moe_fused(const float* __restrict__ X, const float* __restrict__ w1,
               const float* __restrict__ w2, const int* __restrict__ offs,
               const int* __restrict__ toffs, const int* __restrict__ tok,
               const float* __restrict__ cww, float* __restrict__ out) {
    // one buffer, two lives: X chunk as f32 [TM x XSF dwords] during GEMM1 chunks,
    // then h as bf16 [TM x HS ushorts] for GEMM2. (34.3KB vs 49.7KB -> 49.7KB)
    __shared__ unsigned sRaw[(TM * HS) / 2];
    __shared__ int   sTok[TM];
    __shared__ float sCw[TM];
    __shared__ int   sTo[NEXP + 1];

    unsigned*       sXf = sRaw;                    // f32 view
    unsigned short* sH  = (unsigned short*)sRaw;   // bf16 view

    const int tid  = threadIdx.x;
    const int lane = tid & 31;
    const int wave = tid >> 5;

    if (tid <= NEXP) sTo[tid] = toffs[tid];
    __syncthreads();

    const int blk = blockIdx.x;
    if (blk >= sTo[NEXP]) return;
    int e = 0;
    while (blk >= sTo[e + 1]) e++;
    const int tile  = blk - sTo[e];
    const int pbase = offs[e];
    int rows = (offs[e + 1] - pbase) - tile * TM;
    if (rows > TM) rows = TM;
    const int base = pbase + tile * TM;

    if (tid < TM) {
        bool ok  = tid < rows;
        sTok[tid] = tok[ok ? base + tid : base];   // fallback to a valid row
        sCw[tid]  = ok ? cww[base + tid] : 0.f;    // zero weight kills padding rows
    }
    __syncthreads();

    const float* w1e = w1 + (size_t)e * (2 * Idim) * Hdim;
    const float* w2e = w2 + (size_t)e * Hdim * Idim;

    const int arow  = lane & 15;            // A row within 16-row tile
    const int ahalf = (lane >> 4) << 3;     // lanes>=16 start at K+8 (also C-row offset)
    const int bhalf = (lane >> 4) << 3;     // B: lanes>=16 start at K+8 (same as A)
    const int ibase = wave * (GT1 * 16);    // this wave's column base within I

    const unsigned sXfBase = (unsigned)(size_t)(void*)sXf;  // LDS byte offset

    v8f gacc0[GT1] = {}, gacc1[GT1] = {};
    v8f uacc0[GT1] = {}, uacc1[GT1] = {};

    for (int kc = 0; kc < Hdim; kc += KC) {
        // async DMA X[:, kc:kc+KC] (f32) for the TM gathered tokens into LDS
        for (int i = tid; i < TM * (KC / 4); i += NTHREADS) {
            int r  = i / (KC / 4);
            int c4 = (i % (KC / 4)) * 4;
            async_copy16(sXfBase + (unsigned)(r * XSF + c4) * 4u,
                         X + (size_t)sTok[r] * Hdim + kc + c4);
        }
        wait_async0();
        __syncthreads();

        for (int ks = 0; ks < KC; ks += 32) {
            BF a0 = load_a_f32lds(sXf + arow * XSF,        ks + ahalf);  // rows 0..15
            BF a1 = load_a_f32lds(sXf + (16 + arow) * XSF, ks + ahalf);  // rows 16..31
            int kb = kc + ks + bhalf;
#pragma unroll
            for (int t = 0; t < GT1; t++) {
                int n = ibase + t * 16 + (lane & 15);
                BF bg = load_b_f32(w1e + (size_t)n * Hdim + kb);
                gacc0[t] = wmma_bf16(a0, bg, gacc0[t]);
                gacc1[t] = wmma_bf16(a1, bg, gacc1[t]);
                BF bu = load_b_f32(w1e + (size_t)(n + Idim) * Hdim + kb);
                uacc0[t] = wmma_bf16(a0, bu, uacc0[t]);
                uacc1[t] = wmma_bf16(a1, bu, uacc1[t]);
            }
        }
        __syncthreads();   // all waves done with this X chunk before it is overwritten
    }

    // h = silu(g) * u  -> LDS as bf16 [TM x Idim]
#pragma unroll
    for (int t = 0; t < GT1; t++) {
#pragma unroll
        for (int j = 0; j < 8; j++) {
            int m = j + ahalf;
            int c = ibase + t * 16 + (lane & 15);
            float g0 = gacc0[t][j], u0 = uacc0[t][j];
            float g1 = gacc1[t][j], u1 = uacc1[t][j];
            float h0 = (g0 / (1.f + __expf(-g0))) * u0;
            float h1 = (g1 / (1.f + __expf(-g1))) * u1;
            sH[m * HS + c]        = f2bf1(h0);
            sH[(m + 16) * HS + c] = f2bf1(h1);
        }
    }
    __syncthreads();

    // GEMM2: h[TM x I] @ w2e^T -> [TM x H], scaled by combine weight
    v8f oacc0[NT2] = {}, oacc1[NT2] = {};
    const int nbase = wave * (NT2 * 16);
    for (int ks = 0; ks < Idim; ks += 32) {
        BF a0 = load_a_lds(sH + arow * HS,        ks + ahalf);
        BF a1 = load_a_lds(sH + (16 + arow) * HS, ks + ahalf);
        int kb = ks + bhalf;
#pragma unroll
        for (int t = 0; t < NT2; t++) {
            int n = nbase + t * 16 + (lane & 15);
            BF b = load_b_f32(w2e + (size_t)n * Idim + kb);
            oacc0[t] = wmma_bf16(a0, b, oacc0[t]);
            oacc1[t] = wmma_bf16(a1, b, oacc1[t]);
        }
    }
#pragma unroll
    for (int t = 0; t < NT2; t++) {
#pragma unroll
        for (int j = 0; j < 8; j++) {
            int m = j + ahalf;
            int c = nbase + t * 16 + (lane & 15);
            atomicAdd(out + (size_t)sTok[m] * Hdim + c,      oacc0[t][j] * sCw[m]);
            atomicAdd(out + (size_t)sTok[m + 16] * Hdim + c, oacc1[t][j] * sCw[m + 16]);
        }
    }
}

// ---------------- launch ----------------
extern "C" void kernel_launch(void* const* d_in, const int* in_sizes, int n_in,
                              void* d_out, int out_size, void* d_ws, size_t ws_size,
                              hipStream_t stream) {
    (void)n_in; (void)out_size; (void)ws_size;
    const float* X   = (const float*)d_in[0];
    const float* wts = (const float*)d_in[1];
    const int*   ids = (const int*)d_in[2];
    const float* w1  = (const float*)d_in[3];
    const float* w2  = (const float*)d_in[4];
    float* out = (float*)d_out;

    const int T  = in_sizes[0] / Hdim;
    const int NP = T * TOPK;

    int*   cnt   = (int*)d_ws;
    int*   fill  = cnt + NEXP;
    int*   offs  = fill + NEXP;
    int*   toffs = offs + NEXP + 1;
    int*   tok   = toffs + NEXP + 1;
    float* cww   = (float*)(tok + NP);

    hipMemsetAsync(d_ws, 0, 2 * NEXP * sizeof(int), stream);
    hipMemsetAsync(d_out, 0, (size_t)T * Hdim * sizeof(float), stream);

    k_count<<<(NP + 255) / 256, 256, 0, stream>>>(ids, NP, cnt);
    k_scan<<<1, 1, 0, stream>>>(cnt, offs, toffs);
    k_fill<<<(NP + 255) / 256, 256, 0, stream>>>(ids, wts, NP, offs, fill, tok, cww);

    const int maxTiles = (NP + TM - 1) / TM + NEXP;  // upper bound on total tiles
    moe_fused<<<maxTiles, NTHREADS, 0, stream>>>(X, w1, w2, offs, toffs, tok, cww, out);
}
